// RGCN_60559038874083
// MI455X (gfx1250) — matrix operand order repaired and checked by reading, not compiled, not run on hardware
//
#include <hip/hip_runtime.h>
#include <hip/hip_bf16.h>
#include <math.h>

// ---------------------------------------------------------------------------
// Types for CDNA5 WMMA (wave32): 16x16x32 bf16 -> f32
// ---------------------------------------------------------------------------
typedef __attribute__((ext_vector_type(16))) __bf16 v16bf;
typedef __attribute__((ext_vector_type(8)))  float  v8f;

#define LRELU_SLOPE 0.2f

__device__ inline __bf16 f2bf(float f) {
  unsigned u = __float_as_uint(f);
  unsigned r = u + 0x7fffu + ((u >> 16) & 1u);   // round-to-nearest-even
  unsigned short h = (unsigned short)(r >> 16);
  __bf16 b;
  __builtin_memcpy(&b, &h, sizeof(b));
  return b;
}

// ---------------------------------------------------------------------------
// Elementwise helpers
// ---------------------------------------------------------------------------
__global__ void cvt_f32_bf16_kernel(const float* __restrict__ in,
                                    __bf16* __restrict__ out, long n) {
  long i = (long)blockIdx.x * blockDim.x + threadIdx.x;
  long stride = (long)gridDim.x * blockDim.x;
  for (; i < n; i += stride) out[i] = f2bf(in[i]);
}

__global__ void fill_u32_kernel(unsigned* __restrict__ p, unsigned v, long n) {
  long i = (long)blockIdx.x * blockDim.x + threadIdx.x;
  if (i < n) p[i] = v;
}

// H1[i, c] = 2*x[i, c] + b1[0][c] + b1[1][c]   (residual added once per relation)
__global__ void init_h1_kernel(const float* __restrict__ x,
                               const float* __restrict__ b1,
                               float* __restrict__ H1, long n) {
  long i = (long)blockIdx.x * blockDim.x + threadIdx.x;
  if (i >= n) return;
  int c = (int)(i & 255);
  H1[i] = 2.0f * x[i] + b1[c] + b1[256 + c];
}

// out[i, d] = b2[0][d] + b2[1][d]
__global__ void init_out_kernel(const float* __restrict__ b2,
                                float* __restrict__ out, long n) {
  long i = (long)blockIdx.x * blockDim.x + threadIdx.x;
  if (i >= n) return;
  int d = (int)(i & 63);
  out[i] = b2[d] + b2[64 + d];
}

__global__ void add_kernel(float* __restrict__ dst, const float* __restrict__ s, long n) {
  long i = (long)blockIdx.x * blockDim.x + threadIdx.x;
  if (i < n) dst[i] += s[i];
}

__global__ void relu_kernel(float* __restrict__ p, long n) {
  long i = (long)blockIdx.x * blockDim.x + threadIdx.x;
  if (i < n) p[i] = fmaxf(p[i], 0.0f);
}

// h = relu(H1) in place, plus bf16 copy for the layer-2 WMMA GEMMs
__global__ void relu_cvt_kernel(float* __restrict__ H1, __bf16* __restrict__ hb, long n) {
  long i = (long)blockIdx.x * blockDim.x + threadIdx.x;
  if (i >= n) return;
  float v = fmaxf(H1[i], 0.0f);
  H1[i] = v;
  hb[i] = f2bf(v);
}

// ---------------------------------------------------------------------------
// WMMA NT GEMM: C[N,M] = X[N,K] * W[M,K]^T, row-major, bf16 in / f32 out.
// One wave -> 16x64 tile: 1 A-frag, 4 B-frags, 4 accumulators per K-step.
// A-frag layout (ISA 7.12.2, 16-bit A 16x32): lanes 0-15 row m, K {k..k+7, k+16..k+23};
// lanes 16-31 row m, K {k+8..k+15, k+24..k+31}.
// B-frag layout (32x16): lanes 0-15 col n, K {k..k+7, k+16..k+23}; lanes 16-31 col n, K {k+8..k+15, k+24..k+31}.
// ---------------------------------------------------------------------------
__device__ inline v16bf load_a_frag(const __bf16* __restrict__ X, int ld,
                                    int row, int k, int lane) {
  const __bf16* p = X + (long)row * ld + k + ((lane >> 4) << 3);
  v16bf f;
#pragma unroll
  for (int i = 0; i < 8; ++i) { f[i] = p[i]; f[8 + i] = p[16 + i]; }
  return f;
}

__device__ inline v16bf load_b_frag(const __bf16* __restrict__ W, int ld,
                                    int col, int k, int lane) {
  const __bf16* p = W + (long)col * ld + k + ((lane >> 4) << 3);
  v16bf f;
#pragma unroll
  for (int i = 0; i < 8; ++i) { f[i] = p[i]; f[8 + i] = p[16 + i]; }
  return f;
}

__global__ void __launch_bounds__(256)
gemm_nt_bf16_kernel(const __bf16* __restrict__ X, const __bf16* __restrict__ W,
                    float* __restrict__ C, int Nrows, int K, int M) {
  const int lane = threadIdx.x & 31;
  const int wave = threadIdx.x >> 5;
  const int rowTile = blockIdx.x * 8 + wave;    // 16 rows per wave
  const int colBase = blockIdx.y * 64;          // 64 cols per wave
  if (rowTile * 16 >= Nrows) return;            // wave-uniform exit (EXEC stays full)
  const int r  = rowTile * 16 + (lane & 15);
  const int cl = lane & 15;

  v8f c0 = {}, c1 = {}, c2 = {}, c3 = {};
  for (int k = 0; k < K; k += 32) {
    v16bf a  = load_a_frag(X, K, r, k, lane);
    v16bf b0 = load_b_frag(W, K, colBase +  0 + cl, k, lane);
    v16bf b1 = load_b_frag(W, K, colBase + 16 + cl, k, lane);
    v16bf b2 = load_b_frag(W, K, colBase + 32 + cl, k, lane);
    v16bf b3 = load_b_frag(W, K, colBase + 48 + cl, k, lane);
    c0 = __builtin_amdgcn_wmma_f32_16x16x32_bf16(false, a, false, b0, (short)0, c0, false, false);
    c1 = __builtin_amdgcn_wmma_f32_16x16x32_bf16(false, a, false, b1, (short)0, c1, false, false);
    c2 = __builtin_amdgcn_wmma_f32_16x16x32_bf16(false, a, false, b2, (short)0, c2, false, false);
    c3 = __builtin_amdgcn_wmma_f32_16x16x32_bf16(false, a, false, b3, (short)0, c3, false, false);
  }

  // C/D layout: VGPR v -> row v (lanes 0-15) / v+8 (lanes 16-31), col = lane%16
  const int rbase = rowTile * 16 + ((lane >> 4) << 3);
  const int col   = colBase + cl;
#pragma unroll
  for (int v = 0; v < 8; ++v) {
    long o = (long)(rbase + v) * M + col;
    C[o]      = c0[v];
    C[o + 16] = c1[v];
    C[o + 32] = c2[v];
    C[o + 48] = c3[v];
  }
}

// ---------------------------------------------------------------------------
// Edge softmax passes. One wave32 per (edge, head); lanes stride D=64.
// Monotonic uint encoding for float atomic-max (enc > 0 for all finite scores,
// so enc==0 <=> "no in-edges" -> smax := 0, matching the reference's
// where(isfinite(smax), smax, 0)).
// ---------------------------------------------------------------------------
__device__ inline unsigned ord_enc(float f) {
  unsigned u = __float_as_uint(f);
  return (u & 0x80000000u) ? ~u : (u | 0x80000000u);
}

__global__ void edge_score_max_kernel(const float* __restrict__ feat,
                                      const int* __restrict__ src,
                                      const int* __restrict__ dst,
                                      const float* __restrict__ attn,
                                      float* __restrict__ score,
                                      unsigned* __restrict__ smax,
                                      int EH, int H) {
  const int gw   = blockIdx.x * 8 + (threadIdx.x >> 5);
  const int lane = threadIdx.x & 31;
  if (gw >= EH) return;                       // wave-uniform
  const int e = gw / H;
  const int h = gw - e * H;
  const int stride = H << 6;                  // H*64
  const long so = (long)src[e] * stride + h * 64;
  const long dofs = (long)dst[e] * stride + h * 64;
  const float* av = attn + h * 64;

  float acc = 0.0f;
  for (int i = lane; i < 64; i += 32) {
    float v = feat[so + i] + feat[dofs + i];
    v = v > 0.0f ? v : LRELU_SLOPE * v;
    acc += v * av[i];
  }
  for (int off = 16; off > 0; off >>= 1) acc += __shfl_xor(acc, off, 32);
  if (lane == 0) {
    score[gw] = acc;                          // gw == e*H + h
    atomicMax(&smax[(long)dst[e] * H + h], ord_enc(acc));
  }
}

__global__ void decode_smax_kernel(const unsigned* __restrict__ enc,
                                   float* __restrict__ smax, long n) {
  long i = (long)blockIdx.x * blockDim.x + threadIdx.x;
  if (i >= n) return;
  unsigned e = enc[i];
  float f = 0.0f;
  if (e != 0u)
    f = __uint_as_float((e & 0x80000000u) ? (e ^ 0x80000000u) : ~e);
  smax[i] = f;
}

__global__ void edge_exp_kernel(float* __restrict__ score,
                                const int* __restrict__ dst,
                                const float* __restrict__ smax,
                                float* __restrict__ denom,
                                long EH, int H) {
  long i = (long)blockIdx.x * blockDim.x + threadIdx.x;
  if (i >= EH) return;
  long e = i / H;
  int  h = (int)(i - e * H);
  long di = (long)dst[e] * H + h;
  float ex = expf(score[i] - smax[di]);
  score[i] = ex;
  atomicAdd(&denom[di], ex);
}

__global__ void edge_scatter_kernel(const float* __restrict__ feat,
                                    const float* __restrict__ ex,
                                    const float* __restrict__ denom,
                                    const int* __restrict__ src,
                                    const int* __restrict__ dst,
                                    float* __restrict__ acc,
                                    int EH, int H) {
  const int gw   = blockIdx.x * 8 + (threadIdx.x >> 5);
  const int lane = threadIdx.x & 31;
  if (gw >= EH) return;                       // wave-uniform
  const int e = gw / H;
  const int h = gw - e * H;
  const int d0 = dst[e], s0 = src[e];
  const float alpha = ex[gw] / fmaxf(denom[(long)d0 * H + h], 1e-16f);
  const int stride = H << 6;
  const long so = (long)s0 * stride + h * 64;
  const long dofs = (long)d0 * stride + h * 64;
  for (int i = lane; i < 64; i += 32)
    atomicAdd(&acc[dofs + i], alpha * feat[so + i]);
}

// ---------------------------------------------------------------------------
// Host orchestration
// ---------------------------------------------------------------------------
extern "C" void kernel_launch(void* const* d_in, const int* in_sizes, int n_in,
                              void* d_out, int out_size, void* d_ws, size_t ws_size,
                              hipStream_t stream) {
  (void)n_in; (void)out_size; (void)ws_size;
  const float* x     = (const float*)d_in[0];
  const int*   src   = (const int*)  d_in[1];   // (R, E)
  const int*   dst   = (const int*)  d_in[2];   // (R, E)
  const float* W1    = (const float*)d_in[3];   // (R, 256, 256)
  const float* a1    = (const float*)d_in[4];   // (R, 4, 64)
  const float* b1    = (const float*)d_in[5];   // (R, 256)
  const float* W2    = (const float*)d_in[6];   // (R, 64, 256)
  const float* a2    = (const float*)d_in[7];   // (R, 64)
  const float* b2    = (const float*)d_in[8];   // (R, 64)
  const float* Wres2 = (const float*)d_in[9];   // (R, 64, 256)

  const int N = in_sizes[0] / 256;
  const int E = in_sizes[1] / 2;
  const int H = 4;
  const int TPB = 256;

  // ---- workspace carve (relation-sequential reuse keeps feat L2-resident) ----
  char* ws = (char*)d_ws;
  size_t off = 0;
  auto carve = [&](size_t bytes) -> char* {
    char* p = ws + off;
    off += (bytes + 255) & ~(size_t)255;
    return p;
  };
  __bf16*   xb       = (__bf16*)  carve((size_t)N * 256 * 2);   // x in bf16
  __bf16*   hb       = (__bf16*)  carve((size_t)N * 256 * 2);   // h in bf16
  __bf16*   wb       = (__bf16*)  carve((size_t)256 * 256 * 2); // current weight, bf16
  float*    feat     = (float*)   carve((size_t)N * 256 * 4);   // per-relation projection
  float*    H1       = (float*)   carve((size_t)N * 256 * 4);   // layer-1 accumulator / h
  float*    score    = (float*)   carve((size_t)E * H * 4);     // per-edge score / exp
  unsigned* smax_enc = (unsigned*)carve((size_t)N * H * 4);
  float*    smax     = (float*)   carve((size_t)N * H * 4);
  float*    denom    = (float*)   carve((size_t)N * H * 4);
  float*    feat2    = feat;                    // layer-2 reuses feat region
  float*    res2     = feat + (size_t)N * 64;
  float*    out      = (float*)d_out;           // accumulate directly into d_out

  auto blk = [](long n, int t) { return (unsigned)((n + t - 1) / t); };

  const long nXF = (long)N * 256;
  const long nOut = (long)N * 64;
  const int rowTiles = (N + 15) / 16;
  const dim3 gGemm1((rowTiles + 7) / 8, 4);     // M=256 -> 4 column groups
  const dim3 gGemm2((rowTiles + 7) / 8, 1);     // M=64

  // ---- Layer 1 ----
  cvt_f32_bf16_kernel<<<blk(nXF, TPB), TPB, 0, stream>>>(x, xb, nXF);
  init_h1_kernel<<<blk(nXF, TPB), TPB, 0, stream>>>(x, b1, H1, nXF);

  for (int r = 0; r < 2; ++r) {
    const int* sr = src + (long)r * E;
    const int* dr = dst + (long)r * E;

    cvt_f32_bf16_kernel<<<blk(256 * 256, TPB), TPB, 0, stream>>>(
        W1 + (long)r * 256 * 256, wb, 256 * 256);
    gemm_nt_bf16_kernel<<<gGemm1, TPB, 0, stream>>>(xb, wb, feat, N, 256, 256);

    fill_u32_kernel<<<blk((long)N * H, TPB), TPB, 0, stream>>>(smax_enc, 0u, (long)N * H);
    fill_u32_kernel<<<blk((long)N * H, TPB), TPB, 0, stream>>>((unsigned*)denom, 0u, (long)N * H);

    const int EH = E * H;
    edge_score_max_kernel<<<(EH + 7) / 8, TPB, 0, stream>>>(
        feat, sr, dr, a1 + (long)r * H * 64, score, smax_enc, EH, H);
    decode_smax_kernel<<<blk((long)N * H, TPB), TPB, 0, stream>>>(smax_enc, smax, (long)N * H);
    edge_exp_kernel<<<blk((long)EH, TPB), TPB, 0, stream>>>(score, dr, smax, denom, EH, H);
    edge_scatter_kernel<<<(EH + 7) / 8, TPB, 0, stream>>>(
        feat, score, denom, sr, dr, H1, EH, H);
  }

  relu_cvt_kernel<<<blk(nXF, TPB), TPB, 0, stream>>>(H1, hb, nXF);

  // ---- Layer 2 (H=1, D=64) ----
  init_out_kernel<<<blk(nOut, TPB), TPB, 0, stream>>>(b2, out, nOut);

  for (int r = 0; r < 2; ++r) {
    const int* sr = src + (long)r * E;
    const int* dr = dst + (long)r * E;

    cvt_f32_bf16_kernel<<<blk(64 * 256, TPB), TPB, 0, stream>>>(
        W2 + (long)r * 64 * 256, wb, 64 * 256);
    gemm_nt_bf16_kernel<<<gGemm2, TPB, 0, stream>>>(hb, wb, feat2, N, 256, 64);

    cvt_f32_bf16_kernel<<<blk(64 * 256, TPB), TPB, 0, stream>>>(
        Wres2 + (long)r * 64 * 256, wb, 64 * 256);
    gemm_nt_bf16_kernel<<<gGemm2, TPB, 0, stream>>>(hb, wb, res2, N, 256, 64);
    add_kernel<<<blk(nOut, TPB), TPB, 0, stream>>>(out, res2, nOut);

    fill_u32_kernel<<<blk(N, TPB), TPB, 0, stream>>>(smax_enc, 0u, N);
    fill_u32_kernel<<<blk(N, TPB), TPB, 0, stream>>>((unsigned*)denom, 0u, N);

    edge_score_max_kernel<<<(E + 7) / 8, TPB, 0, stream>>>(
        feat2, sr, dr, a2 + (long)r * 64, score, smax_enc, E, 1);
    decode_smax_kernel<<<blk(N, TPB), TPB, 0, stream>>>(smax_enc, smax, N);
    edge_exp_kernel<<<blk(E, TPB), TPB, 0, stream>>>(score, dr, smax, denom, E, 1);
    edge_scatter_kernel<<<(E + 7) / 8, TPB, 0, stream>>>(
        feat2, score, denom, sr, dr, out, E, 1);
  }

  relu_kernel<<<blk(nOut, TPB), TPB, 0, stream>>>(out, nOut);
}
